// FlashMultiHeadAttention_21088289423899
// MI455X (gfx1250) — matrix operand order repaired and checked by reading, not compiled, not run on hardware
//
#include <hip/hip_runtime.h>

typedef __attribute__((ext_vector_type(16))) _Float16 v16h;
typedef __attribute__((ext_vector_type(8)))  _Float16 v8h;
typedef __attribute__((ext_vector_type(4)))  _Float16 v4h;
typedef __attribute__((ext_vector_type(8)))  float    v8f;
typedef __attribute__((ext_vector_type(4)))  float    v4f;

#define D_IN   2048
#define D_OUTD 2048
#define NQKV   6144
#define NHEAD  16
#define DH     128
#define SEQ    2048
#define BATCH  2

static __device__ __forceinline__ v16h cat16(v8h lo, v8h hi) {
    v16h r;
#pragma unroll
    for (int i = 0; i < 8; ++i) { r[i] = lo[i]; r[i + 8] = hi[i]; }
    return r;
}

// group-of-32 LDS swizzle: and=0x1f, or=0, xor=XOR (stays within 16-lane halves for XOR<16)
template <int XOR>
static __device__ __forceinline__ float swz(float v) {
    return __int_as_float(__builtin_amdgcn_ds_swizzle(__float_as_int(v), (XOR << 10) | 0x1f));
}

// CDNA5 async copy: 16B global -> 16B LDS per lane, tracked by ASYNCcnt.
static __device__ __forceinline__ void async_cp16(const void* lds_ptr, const void* gptr) {
    unsigned lds = (unsigned)(unsigned long long)lds_ptr;   // low 32 bits = LDS offset
    asm volatile("global_load_async_to_lds_b128 %0, %1, off"
                 :: "v"(lds), "v"(gptr) : "memory");
}
// Drain so that at most N async ops remain outstanding (in-order completion).
static __device__ __forceinline__ void async_wait10() {
    asm volatile("s_wait_asynccnt 0xa" ::: "memory");
}
static __device__ __forceinline__ void async_wait8() {
    asm volatile("s_wait_asynccnt 0x8" ::: "memory");
}

// ---------------------------------------------------------------------------
// f32 -> f16 elementwise (n multiple of 2048), 8 elements/thread
// ---------------------------------------------------------------------------
__global__ __launch_bounds__(256) void cvt_f16(const float* __restrict__ src,
                                               _Float16* __restrict__ dst) {
    int i = (blockIdx.x * 256 + threadIdx.x) * 8;
    v4f a = *(const v4f*)(src + i);
    v4f b = *(const v4f*)(src + i + 4);
    v8h o;
#pragma unroll
    for (int j = 0; j < 4; ++j) { o[j] = (_Float16)a[j]; o[j + 4] = (_Float16)b[j]; }
    *(v8h*)(dst + i) = o;
}

// ---------------------------------------------------------------------------
// f32 [K][N] -> f16 transposed [N][K], 32x32 tiles via LDS
// ---------------------------------------------------------------------------
__global__ __launch_bounds__(256) void cvt_transpose(const float* __restrict__ src,
                                                     _Float16* __restrict__ dst,
                                                     int K, int N) {
    __shared__ _Float16 t[32][33];
    const int k0 = blockIdx.x * 32, n0 = blockIdx.y * 32;
    const int r  = threadIdx.x >> 3, c4 = (threadIdx.x & 7) * 4;
    v4f f = *(const v4f*)(src + (size_t)(k0 + r) * N + n0 + c4);
#pragma unroll
    for (int i = 0; i < 4; ++i) t[r][c4 + i] = (_Float16)f[i];
    __syncthreads();
    v4h o;
#pragma unroll
    for (int i = 0; i < 4; ++i) o[i] = t[c4 + i][r];
    *(v4h*)(dst + (size_t)(n0 + r) * K + k0 + c4) = o;
}

// ---------------------------------------------------------------------------
// Kernel 1: qkv = xh @ Wqkv_t^T  (f16, double-buffered async-to-LDS staging)
// WG 256 thr (8 waves, 2x4), WG tile 64(M) x 256(N), wave tile 32x64, BK=64.
// Outputs: q,k natural [bh][s][dh];  v transposed vt[bh][dh][s].
// ---------------------------------------------------------------------------
__global__ __launch_bounds__(256) void qkv_gemm(
    const _Float16* __restrict__ A, const _Float16* __restrict__ Bt,
    _Float16* __restrict__ qo, _Float16* __restrict__ ko, _Float16* __restrict__ vt) {
    __shared__ _Float16 ldsA[2][64][72];
    __shared__ _Float16 ldsB[2][256][72];

    const int tid  = threadIdx.x;
    const int wave = tid >> 5, lane = tid & 31;
    const int l16  = lane & 15, hf = lane >> 4;
    const int m0 = blockIdx.x * 64;
    const int n0 = blockIdx.y * 256;
    const int wm = wave >> 2;   // 0..1
    const int wn = wave & 3;    // 0..3

    v8f acc[2][4] = {};

    auto stage = [&](int buf, int kk) {   // 10 async b128 copies per thread
#pragma unroll
        for (int t = 0; t < 2; ++t) {
            int ch = tid * 2 + t;
            int row = ch >> 3, c8 = (ch & 7) * 8;
            async_cp16(&ldsA[buf][row][c8], A + (size_t)(m0 + row) * D_IN + kk + c8);
        }
#pragma unroll
        for (int t = 0; t < 8; ++t) {
            int ch = tid * 8 + t;
            int row = ch >> 3, c8 = (ch & 7) * 8;
            async_cp16(&ldsB[buf][row][c8], Bt + (size_t)(n0 + row) * D_IN + kk + c8);
        }
    };

    stage(0, 0);
    for (int k0 = 0; k0 < D_IN; k0 += 64) {
        const int cur = (k0 >> 6) & 1;
        const int knext = (k0 + 64 < D_IN) ? k0 + 64 : k0;   // clamped: uniform code
        stage(cur ^ 1, knext);
        async_wait10();          // drain current stage's 10 copies; next keeps streaming
        __syncthreads();

        v16h af[2][2];
#pragma unroll
        for (int i = 0; i < 2; ++i)
#pragma unroll
            for (int kc = 0; kc < 2; ++kc) {
                int mr = wm * 32 + i * 16 + l16;
                int cb = kc * 32 + hf * 8;
                af[i][kc] = cat16(*(const v8h*)&ldsA[cur][mr][cb],
                                  *(const v8h*)&ldsA[cur][mr][cb + 16]);
            }
#pragma unroll
        for (int j = 0; j < 4; ++j) {
            v16h bf[2];
#pragma unroll
            for (int kc = 0; kc < 2; ++kc) {
                int nr = wn * 64 + j * 16 + l16;
                int kb = kc * 32 + hf * 8;
                bf[kc] = cat16(*(const v8h*)&ldsB[cur][nr][kb],
                               *(const v8h*)&ldsB[cur][nr][kb + 16]);
            }
#pragma unroll
            for (int i = 0; i < 2; ++i)
#pragma unroll
                for (int kc = 0; kc < 2; ++kc)
                    acc[i][j] = __builtin_amdgcn_wmma_f32_16x16x32_f16(
                        false, af[i][kc], false, bf[kc], (short)0, acc[i][j], false, false);
        }
        __syncthreads();
    }

    // epilogue: scatter q/k (natural), vectorized store for v (transposed)
#pragma unroll
    for (int i = 0; i < 2; ++i) {
#pragma unroll
        for (int j = 0; j < 4; ++j) {
            int ng     = n0 + wn * 64 + j * 16 + l16;
            int which  = ng >> 11;            // 0=q 1=k 2=v
            int within = ng & 2047;
            int h = within >> 7, d = within & 127;
            int sbase = m0 + wm * 32 + i * 16 + hf * 8;     // 8 consecutive s
            int b = sbase >> 11, s = sbase & 2047;
            int bh = b * NHEAD + h;
            if (which == 2) {
                v8h pk;
#pragma unroll
                for (int r = 0; r < 8; ++r) pk[r] = (_Float16)acc[i][j][r];
                *(v8h*)(vt + ((size_t)bh * DH + d) * SEQ + s) = pk;
            } else {
                _Float16* dst = (which == 0) ? qo : ko;
#pragma unroll
                for (int r = 0; r < 8; ++r)
                    dst[((size_t)bh * SEQ + s + r) * DH + d] = (_Float16)acc[i][j][r];
            }
        }
    }
}

// ---------------------------------------------------------------------------
// Kernel 2: causal flash attention.  Grid: (S/128, B*H). 8 waves x 16 q-rows.
// K/V staged with double-buffered async copies (V pre-transposed globally).
// ---------------------------------------------------------------------------
__global__ __launch_bounds__(256) void flash_attn(
    const _Float16* __restrict__ qi, const _Float16* __restrict__ ki,
    const _Float16* __restrict__ vt, _Float16* __restrict__ attn) {
    __shared__ _Float16 ldsK[2][64][136];   // key-major (B-frag for Q.K^T)
    __shared__ _Float16 ldsVt[2][128][72];  // dh-major (B-frag for P.V)
    __shared__ _Float16 ldsP[8][16][72];    // per-wave P relay (D->A layout)

    const int tid  = threadIdx.x;
    const int wave = tid >> 5, lane = tid & 31;
    const int l16  = lane & 15, hf = lane >> 4;
    const int bh = blockIdx.y;
    const int q0 = blockIdx.x * 128;
    const int qw = q0 + wave * 16;
    const float scale = 0.08838834764831845f;  // 1/sqrt(128)

    v16h qa[4];
    {
        const _Float16* qrow = qi + (((size_t)bh) * SEQ + (qw + l16)) * DH;
#pragma unroll
        for (int c = 0; c < 4; ++c)
            qa[c] = cat16(*(const v8h*)(qrow + c * 32 + hf * 8),
                          *(const v8h*)(qrow + c * 32 + hf * 8 + 16));
    }

    v8f oacc[8] = {};
    float mrow[8], lrow[8];
#pragma unroll
    for (int r = 0; r < 8; ++r) { mrow[r] = -1e30f; lrow[r] = 0.f; }

    const int kbmax = (q0 + 127) >> 6;

    auto stage = [&](int buf, int kb) {    // 8 async b128 copies per thread
        const size_t kbase = (((size_t)bh) * SEQ + kb * 64) * DH;
        const size_t vbase = ((size_t)bh) * DH * SEQ + kb * 64;
#pragma unroll
        for (int t = 0; t < 4; ++t) {
            int ch = tid * 4 + t;
            int krow = ch >> 4, kc8 = (ch & 15) * 8;
            async_cp16(&ldsK[buf][krow][kc8], ki + kbase + (size_t)krow * DH + kc8);
            int d = ch >> 3, s8 = (ch & 7) * 8;
            async_cp16(&ldsVt[buf][d][s8], vt + vbase + (size_t)d * SEQ + s8);
        }
    };

    stage(0, 0);
    for (int kb = 0; kb <= kbmax; ++kb) {
        const int cur = kb & 1;
        stage(cur ^ 1, (kb < kbmax) ? kb + 1 : kb);   // clamped: uniform
        async_wait8();
        __syncthreads();

        // S = Q K^T (scaled, causal-masked)
        float sv[4][8];
#pragma unroll
        for (int nt = 0; nt < 4; ++nt) {
            v8f s = {};
#pragma unroll
            for (int c = 0; c < 4; ++c) {
                int krow = nt * 16 + l16;
                int kcol = c * 32 + hf * 8;
                v16h bf = cat16(*(const v8h*)&ldsK[cur][krow][kcol],
                                *(const v8h*)&ldsK[cur][krow][kcol + 16]);
                s = __builtin_amdgcn_wmma_f32_16x16x32_f16(
                    false, qa[c], false, bf, (short)0, s, false, false);
            }
            int colg = kb * 64 + nt * 16 + l16;
#pragma unroll
            for (int r = 0; r < 8; ++r) {
                int rowg = qw + r + 8 * hf;
                sv[nt][r] = (colg <= rowg) ? s[r] * scale : -1e30f;
            }
        }

        // online softmax (row r+8*hf; its 16 cols live across the 16-lane half)
#pragma unroll
        for (int r = 0; r < 8; ++r) {
            float mx = fmaxf(fmaxf(sv[0][r], sv[1][r]), fmaxf(sv[2][r], sv[3][r]));
            mx = fmaxf(mx, swz<1>(mx)); mx = fmaxf(mx, swz<2>(mx));
            mx = fmaxf(mx, swz<4>(mx)); mx = fmaxf(mx, swz<8>(mx));
            float mnew  = fmaxf(mrow[r], mx);
            float alpha = __expf(mrow[r] - mnew);
            float psum = 0.f;
#pragma unroll
            for (int nt = 0; nt < 4; ++nt) {
                float p = __expf(sv[nt][r] - mnew);
                sv[nt][r] = p;
                psum += p;
            }
            psum += swz<1>(psum); psum += swz<2>(psum);
            psum += swz<4>(psum); psum += swz<8>(psum);
            lrow[r] = lrow[r] * alpha + psum;
            mrow[r] = mnew;
#pragma unroll
            for (int d = 0; d < 8; ++d) oacc[d][r] *= alpha;
        }

        // relay P through LDS: D-layout -> A-layout
#pragma unroll
        for (int nt = 0; nt < 4; ++nt)
#pragma unroll
            for (int r = 0; r < 8; ++r)
                ldsP[wave][r + 8 * hf][nt * 16 + l16] = (_Float16)sv[nt][r];
        __syncthreads();

        // O += P V
        v16h pa[2];
#pragma unroll
        for (int c2 = 0; c2 < 2; ++c2) {
            int base = c2 * 32 + hf * 8;
            pa[c2] = cat16(*(const v8h*)&ldsP[wave][l16][base],
                           *(const v8h*)&ldsP[wave][l16][base + 16]);
        }
#pragma unroll
        for (int d = 0; d < 8; ++d) {
#pragma unroll
            for (int c2 = 0; c2 < 2; ++c2) {
                int vrow = d * 16 + l16;
                int vcol = c2 * 32 + hf * 8;
                v16h vb = cat16(*(const v8h*)&ldsVt[cur][vrow][vcol],
                                *(const v8h*)&ldsVt[cur][vrow][vcol + 16]);
                oacc[d] = __builtin_amdgcn_wmma_f32_16x16x32_f16(
                    false, pa[c2], false, vb, (short)0, oacc[d], false, false);
            }
        }
        __syncthreads();
    }

    // normalize + write attn [B, S, D_OUT] (col = h*128 + d) as f16
    const int b = bh >> 4, h = bh & 15;
#pragma unroll
    for (int r = 0; r < 8; ++r) {
        float inv = 1.0f / lrow[r];
        int sg = qw + r + 8 * hf;
        _Float16* dst = attn + ((size_t)b * SEQ + sg) * D_OUTD + h * DH;
#pragma unroll
        for (int d = 0; d < 8; ++d)
            dst[d * 16 + l16] = (_Float16)(oacc[d][r] * inv);
    }
}

// ---------------------------------------------------------------------------
// Kernel 3: out = attn @ Wproj_t^T + bias (f16 WMMA, f32 out), double-buffered
// ---------------------------------------------------------------------------
__global__ __launch_bounds__(256) void proj_gemm(
    const _Float16* __restrict__ A, const _Float16* __restrict__ Bt,
    const float* __restrict__ bias, float* __restrict__ out) {
    __shared__ _Float16 ldsA[2][64][72];
    __shared__ _Float16 ldsB[2][256][72];

    const int tid  = threadIdx.x;
    const int wave = tid >> 5, lane = tid & 31;
    const int l16  = lane & 15, hf = lane >> 4;
    const int m0 = blockIdx.x * 64;
    const int n0 = blockIdx.y * 256;
    const int wm = wave >> 2, wn = wave & 3;

    v8f acc[2][4] = {};

    auto stage = [&](int buf, int kk) {
#pragma unroll
        for (int t = 0; t < 2; ++t) {
            int ch = tid * 2 + t;
            int row = ch >> 3, c8 = (ch & 7) * 8;
            async_cp16(&ldsA[buf][row][c8], A + (size_t)(m0 + row) * D_OUTD + kk + c8);
        }
#pragma unroll
        for (int t = 0; t < 8; ++t) {
            int ch = tid * 8 + t;
            int row = ch >> 3, c8 = (ch & 7) * 8;
            async_cp16(&ldsB[buf][row][c8], Bt + (size_t)(n0 + row) * D_OUTD + kk + c8);
        }
    };

    stage(0, 0);
    for (int k0 = 0; k0 < D_OUTD; k0 += 64) {
        const int cur = (k0 >> 6) & 1;
        const int knext = (k0 + 64 < D_OUTD) ? k0 + 64 : k0;
        stage(cur ^ 1, knext);
        async_wait10();
        __syncthreads();

        v16h af[2][2];
#pragma unroll
        for (int i = 0; i < 2; ++i)
#pragma unroll
            for (int kc = 0; kc < 2; ++kc) {
                int mr = wm * 32 + i * 16 + l16;
                int cb = kc * 32 + hf * 8;
                af[i][kc] = cat16(*(const v8h*)&ldsA[cur][mr][cb],
                                  *(const v8h*)&ldsA[cur][mr][cb + 16]);
            }
#pragma unroll
        for (int j = 0; j < 4; ++j) {
            v16h bf[2];
#pragma unroll
            for (int kc = 0; kc < 2; ++kc) {
                int nr = wn * 64 + j * 16 + l16;
                int kb = kc * 32 + hf * 8;
                bf[kc] = cat16(*(const v8h*)&ldsB[cur][nr][kb],
                               *(const v8h*)&ldsB[cur][nr][kb + 16]);
            }
#pragma unroll
            for (int i = 0; i < 2; ++i)
#pragma unroll
                for (int kc = 0; kc < 2; ++kc)
                    acc[i][j] = __builtin_amdgcn_wmma_f32_16x16x32_f16(
                        false, af[i][kc], false, bf[kc], (short)0, acc[i][j], false, false);
        }
        __syncthreads();
    }

#pragma unroll
    for (int i = 0; i < 2; ++i) {
#pragma unroll
        for (int j = 0; j < 4; ++j) {
            int ng = n0 + wn * 64 + j * 16 + l16;
            float bv = bias[ng];
#pragma unroll
            for (int r = 0; r < 8; ++r) {
                int mg = m0 + wm * 32 + i * 16 + r + hf * 8;
                out[(size_t)mg * D_OUTD + ng] = acc[i][j][r] + bv;
            }
        }
    }
}

extern "C" void kernel_launch(void* const* d_in, const int* in_sizes, int n_in,
                              void* d_out, int out_size, void* d_ws, size_t ws_size,
                              hipStream_t stream) {
    (void)in_sizes; (void)n_in; (void)out_size; (void)ws_size;
    const float* x     = (const float*)d_in[0];
    const float* Wqkv  = (const float*)d_in[1];
    const float* Wproj = (const float*)d_in[2];
    const float* bproj = (const float*)d_in[3];
    float* out = (float*)d_out;

    _Float16* xh     = (_Float16*)d_ws;                       // [4096][2048]
    _Float16* wqkvt  = xh     + (size_t)4096 * D_IN;          // [6144][2048]
    _Float16* wprojt = wqkvt  + (size_t)NQKV * D_IN;          // [2048][2048]
    _Float16* q      = wprojt + (size_t)D_OUTD * D_OUTD;      // [bh][s][dh]
    _Float16* k      = q      + (size_t)BATCH * NHEAD * SEQ * DH;
    _Float16* vt     = k      + (size_t)BATCH * NHEAD * SEQ * DH;  // [bh][dh][s]
    _Float16* attn   = vt     + (size_t)BATCH * NHEAD * SEQ * DH;  // [4096][2048]

    cvt_f16      <<<dim3((4096 * D_IN) / 2048), 256, 0, stream>>>(x, xh);
    cvt_transpose<<<dim3(D_IN / 32, NQKV / 32), 256, 0, stream>>>(Wqkv, wqkvt, D_IN, NQKV);
    cvt_transpose<<<dim3(D_OUTD / 32, D_OUTD / 32), 256, 0, stream>>>(Wproj, wprojt, D_OUTD, D_OUTD);

    qkv_gemm  <<<dim3(4096 / 64, NQKV / 256), 256, 0, stream>>>(xh, wqkvt, q, k, vt);
    flash_attn<<<dim3(SEQ / 128, BATCH * NHEAD), 256, 0, stream>>>(q, k, vt, attn);
    proj_gemm <<<dim3(4096 / 64, D_OUTD / 256), 256, 0, stream>>>(attn, wprojt, bproj, out);
}
